// Attention_14714557956326
// MI455X (gfx1250) — matrix operand order repaired and checked by reading, not compiled, not run on hardware
//
#include <hip/hip_runtime.h>

// ---------------------------------------------------------------------------
// Attention with relative positional bias, CDNA5 (gfx1250) WMMA implementation
//   b=4, n=1024, heads=16, dim_head=64, dim=1024, max_pos=512
// All matmuls use v_wmma_f32_16x16x32_f16 (f16 operands, f32 accumulate).
// Workspace layout (assumes ws_size >= ~40MB):
//   xh[4M] Wqt[1M] Wkvt[2M] qh[4M] kh[4M] vth[4M] ph[65600]   (all _Float16)
// ---------------------------------------------------------------------------

typedef __attribute__((ext_vector_type(16))) _Float16 v16h;
typedef __attribute__((ext_vector_type(8)))  float    v8f;

#define BATCH   4
#define HEADS   16
#define DHEAD   64
#define NSEQ    1024
#define DIM     1024
#define NPOS    1025   // 2*512 + 1

union V16U { unsigned u[8]; v16h h; };

// A-matrix 16x32 f16 layout (ISA 7.12.2): lane m = L%16, hi = L/16.
// VGPR v<4:  K = 8*hi + 2v, +1 ; VGPR v>=4: K = 16 + 8*hi + 2(v-4), +1
// => u32 words:  p[4*hi + v] (v<4),  p[8 + 4*hi + (v-4)]
__device__ __forceinline__ v16h loadA16x32(const _Float16* rowPtr, int hi) {
    const unsigned* p = (const unsigned*)rowPtr;
    V16U r;
    int b0 = 4 * hi;
#pragma unroll
    for (int v = 0; v < 4; ++v) r.u[v] = p[b0 + v];
#pragma unroll
    for (int v = 0; v < 4; ++v) r.u[4 + v] = p[8 + b0 + v];
    return r.h;
}

// B-matrix 32x16 f16 layout: lane n = L%16, same K order as A:
// VGPR v<4: K = 8*hi + 2v, +1 ; VGPR v>=4: K = 16 + 8*hi + 2(v-4), +1
// colPtr points at Bt[n][k0] (Bt = B transposed, row-major [N][K]).
__device__ __forceinline__ v16h loadB32x16(const _Float16* colPtr, int hi) {
    const unsigned* p = (const unsigned*)colPtr;
    V16U r;
    int b0 = 4 * hi;
#pragma unroll
    for (int v = 0; v < 8; ++v) r.u[v] = p[(v < 4 ? b0 + v : 8 + b0 + v - 4)];
    return r.h;
}

__device__ __forceinline__ v8f wmma_f16(v16h a, v16h b, v8f c) {
    return __builtin_amdgcn_wmma_f32_16x16x32_f16(false, a, false, b,
                                                  (short)0, c, false, false);
}

__device__ __forceinline__ float half16_max(float v) {
    v = fmaxf(v, __shfl_xor(v, 1, 32));
    v = fmaxf(v, __shfl_xor(v, 2, 32));
    v = fmaxf(v, __shfl_xor(v, 4, 32));
    v = fmaxf(v, __shfl_xor(v, 8, 32));
    return v;
}
__device__ __forceinline__ float half16_sum(float v) {
    v += __shfl_xor(v, 1, 32);
    v += __shfl_xor(v, 2, 32);
    v += __shfl_xor(v, 4, 32);
    v += __shfl_xor(v, 8, 32);
    return v;
}

// ---------------------------------------------------------------------------
// Prep kernels
// ---------------------------------------------------------------------------
__global__ void cast_f32_to_f16(const float* __restrict__ x,
                                _Float16* __restrict__ y, int n) {
    int i = blockIdx.x * blockDim.x + threadIdx.x;
    if (i < n) y[i] = (_Float16)x[i];
}

// y[c*K + k] = (f16) x[k*N + c]   (x is [K][N] row-major)
__global__ void transpose_cast(const float* __restrict__ x,
                               _Float16* __restrict__ y, int K, int N) {
    int idx = blockIdx.x * blockDim.x + threadIdx.x;
    if (idx < K * N) {
        int k = idx % K;
        int c = idx / K;
        y[idx] = (_Float16)x[(size_t)k * N + c];
    }
}

// ---------------------------------------------------------------------------
// Projection GEMM: C[M=4096, Nt] = A[M,1024] * B,  B given transposed [Nt][1024]
// One wave computes a 16x64 output tile (4 col-tiles, shared A regs).
// MODE 0: write Q (scaled by 1/8) to qout[b,h,i,d]
// MODE 1: cols<1024 -> K to kout[b,h,i,d]; cols>=1024 -> V to vtout[b,h,d,i]
// ---------------------------------------------------------------------------
template <int MODE>
__global__ __launch_bounds__(256) void gemm_proj(
    const _Float16* __restrict__ A, const _Float16* __restrict__ Bt,
    _Float16* __restrict__ qout, _Float16* __restrict__ kout,
    _Float16* __restrict__ vtout, int Ntiles4) {
    int wv   = blockIdx.x * (blockDim.x >> 5) + (threadIdx.x >> 5);
    int lane = threadIdx.x & 31;
    int nloc = lane & 15;
    int hi   = lane >> 4;
    int tM   = wv / Ntiles4;
    int tN4  = wv % Ntiles4;

    const _Float16* arow = A + (size_t)(tM * 16 + nloc) * DIM;
    v8f acc[4] = {};
    for (int k0 = 0; k0 < DIM; k0 += 32) {
        v16h a = loadA16x32(arow + k0, hi);
#pragma unroll
        for (int t = 0; t < 4; ++t) {
            const _Float16* bcol =
                Bt + (size_t)(tN4 * 64 + t * 16 + nloc) * DIM + k0;
            acc[t] = wmma_f16(a, loadB32x16(bcol, hi), acc[t]);
        }
    }
    // Epilogue: C layout element (m = r + 8*hi, ncol = lane%16)
    int gmBase = tM * 16;
#pragma unroll
    for (int t = 0; t < 4; ++t) {
        int c = tN4 * 64 + t * 16 + nloc;
#pragma unroll
        for (int r = 0; r < 8; ++r) {
            int gm = gmBase + r + 8 * hi;
            int b = gm >> 10, i = gm & 1023;
            float val = acc[t][r];
            if (MODE == 0) {
                int h = c >> 6, d = c & 63;
                qout[(((size_t)(b * HEADS + h) * NSEQ) + i) * DHEAD + d] =
                    (_Float16)(val * 0.125f);  // fold scale = dhead^-0.5
            } else {
                if (c < 1024) {
                    int h = c >> 6, d = c & 63;
                    kout[(((size_t)(b * HEADS + h) * NSEQ) + i) * DHEAD + d] =
                        (_Float16)val;
                } else {
                    int c2 = c - 1024;
                    int h = c2 >> 6, d = c2 & 63;
                    vtout[(((size_t)(b * HEADS + h) * DHEAD) + d) * NSEQ + i] =
                        (_Float16)val;
                }
            }
        }
    }
}

// ---------------------------------------------------------------------------
// Fused attention: one wave per (b, h, 16-row query block).
// Phase 1: S_pos = Qblk @ pos_emb^T  (16 x 1025, f16) into LDS via WMMA.
// Phase 2: flash loop over j in steps of 32: S = Qblk@K^T + bias, online
//          softmax, P bounced through LDS (C-layout -> A-layout), O += P@V.
// ---------------------------------------------------------------------------
__global__ __launch_bounds__(32) void attn_kernel(
    const _Float16* __restrict__ qh, const _Float16* __restrict__ kh,
    const _Float16* __restrict__ vth, const _Float16* __restrict__ ph,
    float* __restrict__ out) {
    __shared__ _Float16 spos[16][1040];  // pos scores for this row block
    __shared__ _Float16 pbuf[16][36];    // P tile transpose staging

    int wid  = blockIdx.x;    // 0..4095
    int bh   = wid >> 6;      // b*16 + h
    int iblk = wid & 63;
    int lane = threadIdx.x;
    int nloc = lane & 15;
    int hi   = lane >> 4;

    const _Float16* Qb = qh + (size_t)bh * NSEQ * DHEAD;
    const _Float16* Kb = kh + (size_t)bh * NSEQ * DHEAD;
    const _Float16* Vt = vth + (size_t)bh * DHEAD * NSEQ;

    // Q block (16x64) in A-layout, two K=32 chunks, held in regs for all GEMMs
    const _Float16* qrow = Qb + (size_t)(iblk * 16 + nloc) * DHEAD;
    v16h qa0 = loadA16x32(qrow + 0, hi);
    v16h qa1 = loadA16x32(qrow + 32, hi);

    // ---- Phase 1: positional scores into LDS ----
    for (int pt = 0; pt < 65; ++pt) {
        int p  = pt * 16 + nloc;
        int pe = p > (NPOS - 1) ? (NPOS - 1) : p;
        const _Float16* prow = ph + (size_t)pe * DHEAD;
        v8f acc = {};
        acc = wmma_f16(qa0, loadB32x16(prow + 0, hi), acc);
        acc = wmma_f16(qa1, loadB32x16(prow + 32, hi), acc);
#pragma unroll
        for (int r = 0; r < 8; ++r) {
            int m = r + 8 * hi;
            if (p <= NPOS - 1) spos[m][p] = (_Float16)acc[r];
        }
    }
    asm volatile("s_wait_dscnt 0" ::: "memory");

    // ---- Phase 2: flash attention loop ----
    float mrow[8], lrow[8];
#pragma unroll
    for (int r = 0; r < 8; ++r) { mrow[r] = -1e30f; lrow[r] = 0.0f; }
    v8f O[4] = {};

    int ig0 = iblk * 16;
    for (int jt = 0; jt < NSEQ / 32; ++jt) {
        int j0 = jt * 32;
        // scores: two 16-wide column tiles, K = 64 (two chained WMMAs each)
        const _Float16* k0p = Kb + (size_t)(j0 + nloc) * DHEAD;
        const _Float16* k1p = Kb + (size_t)(j0 + 16 + nloc) * DHEAD;
        v8f S0 = {}, S1 = {};
        S0 = wmma_f16(qa0, loadB32x16(k0p + 0, hi), S0);
        S0 = wmma_f16(qa1, loadB32x16(k0p + 32, hi), S0);
        S1 = wmma_f16(qa0, loadB32x16(k1p + 0, hi), S1);
        S1 = wmma_f16(qa1, loadB32x16(k1p + 32, hi), S1);

        // add relative positional bias: p = clip(i - j, +-512) + 512
#pragma unroll
        for (int r = 0; r < 8; ++r) {
            int m  = r + 8 * hi;
            int ig = ig0 + m;
            int d0 = ig - (j0 + nloc);
            d0 = d0 < -512 ? -512 : (d0 > 512 ? 512 : d0);
            S0[r] += (float)spos[m][d0 + 512];
            int d1 = ig - (j0 + 16 + nloc);
            d1 = d1 < -512 ? -512 : (d1 > 512 ? 512 : d1);
            S1[r] += (float)spos[m][d1 + 512];
        }

        // online softmax update (row m lives on the 16 lanes sharing `hi`)
        float alpha[8];
#pragma unroll
        for (int r = 0; r < 8; ++r) {
            float v = half16_max(fmaxf(S0[r], S1[r]));
            float mnew = fmaxf(mrow[r], v);
            alpha[r] = __expf(mrow[r] - mnew);
            mrow[r] = mnew;
        }
#pragma unroll
        for (int r = 0; r < 8; ++r) {
            float p0 = __expf(S0[r] - mrow[r]);
            float p1 = __expf(S1[r] - mrow[r]);
            lrow[r] = lrow[r] * alpha[r] + half16_sum(p0 + p1);
            int m = r + 8 * hi;
            pbuf[m][nloc]      = (_Float16)p0;
            pbuf[m][16 + nloc] = (_Float16)p1;
        }
        asm volatile("s_wait_dscnt 0" ::: "memory");
        v16h pa = loadA16x32(&pbuf[nloc][0], hi);  // P (16x32) in A-layout

        // O = O*alpha + P @ V(j0..j0+31, :)
#pragma unroll
        for (int t = 0; t < 4; ++t) {
#pragma unroll
            for (int r = 0; r < 8; ++r) O[t][r] *= alpha[r];
            const _Float16* vcol = Vt + (size_t)(t * 16 + nloc) * NSEQ + j0;
            O[t] = wmma_f16(pa, loadB32x16(vcol, hi), O[t]);
        }
    }

    // normalize and store: out[b,h, i, d]  f32
    float* ob = out + (size_t)bh * NSEQ * DHEAD + (size_t)ig0 * DHEAD;
#pragma unroll
    for (int t = 0; t < 4; ++t) {
#pragma unroll
        for (int r = 0; r < 8; ++r) {
            int m = r + 8 * hi;
            ob[(size_t)m * DHEAD + t * 16 + nloc] = O[t][r] / lrow[r];
        }
    }
}

// ---------------------------------------------------------------------------
extern "C" void kernel_launch(void* const* d_in, const int* in_sizes, int n_in,
                              void* d_out, int out_size, void* d_ws,
                              size_t ws_size, hipStream_t stream) {
    const float* x   = (const float*)d_in[0];  // [4,1024,1024]
    const float* Wq  = (const float*)d_in[1];  // [1024,1024]
    const float* Wkv = (const float*)d_in[2];  // [1024,2048]
    const float* pe  = (const float*)d_in[3];  // [1025,64]
    float* out       = (float*)d_out;          // [4,16,1024,64]

    size_t off = 0;
    auto carve = [&](size_t halfs) {
        _Float16* p = (_Float16*)((char*)d_ws + off);
        off += ((halfs * 2 + 255) / 256) * 256;
        return p;
    };
    _Float16* xh   = carve((size_t)BATCH * NSEQ * DIM);   // 4M
    _Float16* Wqt  = carve((size_t)DIM * DIM);            // 1M  [N][K]
    _Float16* Wkvt = carve((size_t)2 * DIM * DIM);        // 2M  [N][K]
    _Float16* qhp  = carve((size_t)BATCH * HEADS * NSEQ * DHEAD);
    _Float16* khp  = carve((size_t)BATCH * HEADS * NSEQ * DHEAD);
    _Float16* vtp  = carve((size_t)BATCH * HEADS * DHEAD * NSEQ);
    _Float16* php  = carve((size_t)NPOS * DHEAD);

    int nx = BATCH * NSEQ * DIM;
    cast_f32_to_f16<<<(nx + 255) / 256, 256, 0, stream>>>(x, xh, nx);
    transpose_cast<<<(DIM * DIM + 255) / 256, 256, 0, stream>>>(Wq, Wqt, DIM, DIM);
    transpose_cast<<<(2 * DIM * DIM + 255) / 256, 256, 0, stream>>>(Wkv, Wkvt, DIM, 2 * DIM);
    int np = NPOS * DHEAD;
    cast_f32_to_f16<<<(np + 255) / 256, 256, 0, stream>>>(pe, php, np);

    // Q projection: 4096x1024, 4096 waves of 16x64 tiles -> 512 blocks x 8 waves
    gemm_proj<0><<<512, 256, 0, stream>>>(xh, Wqt, qhp, nullptr, nullptr, 16);
    // KV projection: 4096x2048 -> 8192 waves -> 1024 blocks
    gemm_proj<1><<<1024, 256, 0, stream>>>(xh, Wkvt, nullptr, khp, vtp, 32);

    // fused attention: (b*h) * (n/16) = 64 * 64 = 4096 waves, one per block
    attn_kernel<<<BATCH * HEADS * (NSEQ / 16), 32, 0, stream>>>(qhp, khp, vtp,
                                                                php, out);
}
